// LossEvaluator_33406255629018
// MI455X (gfx1250) — matrix operand order repaired and checked by reading, not compiled, not run on hardware
//
#include <hip/hip_runtime.h>
#include <hip/hip_bf16.h>

#define SEQ    64
#define B_     128
#define NS     64
#define NC     4
#define INDIM  64
#define ZDIM   32
#define HIDN   128
#define MROWS  64      // rows per workgroup: 1 batch column x 64 samples
#define NTHREADS 128   // 4 waves of 32
#define LOG2PI_ 1.8378770664093453f

typedef __attribute__((ext_vector_type(16))) _Float16 v16h;
typedef __attribute__((ext_vector_type(8)))  float    v8f;

union FragA { _Float16 e[16]; v16h h; };
union FragB { unsigned int u[8]; v16h h; };
union H2U  { _Float16 h2[2]; unsigned int u; };

// ---------------- LDS layout (byte offsets into dynamic shared) ----------------
#define WG1_OFF   0                        // 3*8 frags * 1024B = 24576
#define WM1_OFF   (WG1_OFF + 24576)
#define WG2_OFF   (WM1_OFF + 24576)        // 4*2 frags = 8192
#define WM2_OFF   (WG2_OFF + 8192)
#define WLIN_OFF  (WM2_OFF + 8192)         // 3*2 frags = 6144
#define WSC_OFF   (WLIN_OFF + 6144)        // 1*2 frags = 2048
#define WE_OFF    (WSC_OFF + 2048)         // 1 frag    = 1024
#define ZPREV_OFF (WE_OFF + 1024)          // 64*32*4   = 8192
#define H1_OFF    (ZPREV_OFF + 8192)       // 64*128*4  = 32768
#define GATE_OFF  (H1_OFF + 32768)         // 8192
#define PROP_OFF  (GATE_OFF + 8192)        // 8192
#define LOC_OFF   (PROP_OFF + 8192)        // 8192
#define SCL_OFF   (LOC_OFF + 8192)         // 8192
#define OUTB_OFF  (SCL_OFF + 8192)         // 64*16*4   = 4096
#define RED_OFF   (OUTB_OFF + 4096)        // 4*64*4    = 1024
#define BASE_OFF  (RED_OFF + 1024)         // 4*4
#define SMEM_BYTES (BASE_OFF + 16)

// ---------------- helpers ----------------
__device__ __forceinline__ v8f wmma_f16(v16h a, v16h b, v8f c) {
  return __builtin_amdgcn_wmma_f32_16x16x32_f16(false, a, false, b, (short)0, c, false, false);
}

// A-fragment (16x32 f16, ISA 7.12.2 layout) packed from a row-major f32 LDS buffer
__device__ __forceinline__ v16h packA_lds(const float* src, int stride, int rbase,
                                          int kbase, int lane, bool relu) {
  FragA fa;
  const float* rowp = src + (rbase + (lane & 15)) * stride + kbase;
  const int khi = (lane >> 4) << 3; // 0 or 8
#pragma unroll
  for (int e = 0; e < 16; ++e) {
    int k = (((e >> 1) & 3) << 1) + (e & 1) + khi + ((e >> 3) << 4);
    float v = rowp[k];
    if (relu) v = fmaxf(v, 0.f);
    fa.e[e] = (_Float16)v;
  }
  return fa.h;
}

// A-fragment with all 16 rows identical (the broadcast x part of h)
__device__ __forceinline__ v16h packA_bc(const float* xv, int kbase, int lane) {
  FragA fa;
  const int khi = (lane >> 4) << 3;
#pragma unroll
  for (int e = 0; e < 16; ++e) {
    int k = (((e >> 1) & 3) << 1) + (e & 1) + khi + ((e >> 3) << 4);
    fa.e[e] = (_Float16)xv[kbase + k];
  }
  return fa.h;
}

// B-fragment: 8 consecutive dwords per lane from pre-packed LDS
__device__ __forceinline__ v16h loadB(const unsigned int* fr, int frag, int lane) {
  FragB fb;
  const unsigned int* p = fr + frag * 256 + lane * 8;
#pragma unroll
  for (int i = 0; i < 8; ++i) fb.u[i] = p[i];
  return fb.h;
}

// Pack a K x N row-major f32 weight into B-fragments (f16 pairs), frag = kt*ntiles+nt
__device__ void pack_weight(const float* W, int K, int N, unsigned int* dst,
                            int tid, int nth) {
  const int ntiles = N >> 4;
  const int total = (K >> 5) * ntiles * 256;
  for (int idx = tid; idx < total; idx += nth) {
    int frag = idx >> 8;
    int slot = idx & 255;
    int lane = slot >> 3;
    int v = slot & 7;
    int kt = frag / ntiles, nt = frag - kt * ntiles;
    int n = (nt << 4) + (lane & 15);
    int k0 = (kt << 5) + ((lane >> 4) << 3) + (v << 1) + ((v >> 2) << 3);
    H2U p;
    p.h2[0] = (_Float16)W[k0 * N + n];
    p.h2[1] = (_Float16)W[(k0 + 1) * N + n];
    dst[idx] = p.u;
  }
}

__device__ __forceinline__ unsigned int rotl32(unsigned int x, int n) {
  return (x << n) | (x >> (32 - n));
}
// Threefry-2x32, 20 rounds (JAX PRNG core)
__device__ __forceinline__ void threefry2x32(unsigned int k0, unsigned int k1,
                                             unsigned int c0, unsigned int c1,
                                             unsigned int& o0, unsigned int& o1) {
  unsigned int ks0 = k0, ks1 = k1, ks2 = 0x1BD11BDAu ^ k0 ^ k1;
  unsigned int x0 = c0 + ks0, x1 = c1 + ks1;
  const int ra[8] = {13, 15, 26, 6, 17, 29, 16, 24};
#pragma unroll
  for (int g = 0; g < 5; ++g) {
#pragma unroll
    for (int j = 0; j < 4; ++j) {
      x0 += x1;
      x1 = rotl32(x1, ra[((g & 1) << 2) + j]);
      x1 ^= x0;
    }
    unsigned int i = g + 1;
    unsigned int m0 = i % 3, m1 = (i + 1) % 3;
    x0 += (m0 == 0) ? ks0 : ((m0 == 1) ? ks1 : ks2);
    x1 += ((m1 == 0) ? ks0 : ((m1 == 1) ? ks1 : ks2)) + i;
  }
  o0 = x0; o1 = x1;
}
__device__ __forceinline__ float u01(unsigned int b) {
  return (float)(b >> 8) * (1.0f / 16777216.0f);
}
__device__ __forceinline__ float bern_lp(float s, float p) {
  p = fminf(fmaxf(p, 1e-6f), 1.f - 1e-6f);
  return (s > 0.5f) ? logf(p) : log1pf(-p);
}

// h(M x 96) @ W(96 x 128) + bias, relu, -> dst (row-major 128-col f32)
__device__ __forceinline__ void gemm_h_to_hid(v16h a0, v16h a1, v16h a2,
                                              const unsigned int* wf,
                                              const float* bias, float* dst,
                                              int rbase, int lane) {
#pragma unroll
  for (int nt = 0; nt < 8; ++nt) {
    v8f c = {};
    c = wmma_f16(a0, loadB(wf, nt, lane), c);
    c = wmma_f16(a1, loadB(wf, 8 + nt, lane), c);
    c = wmma_f16(a2, loadB(wf, 16 + nt, lane), c);
    int nl = lane & 15, mh = (lane >> 4) << 3;
    float bv = bias[nt * 16 + nl];
#pragma unroll
    for (int v = 0; v < 8; ++v)
      dst[(rbase + mh + v) * HIDN + nt * 16 + nl] = fmaxf(c[v] + bv, 0.f);
  }
}

// src(M x 128) @ W(128 x 32) + bias -> dst (32-col). mode 0: raw, 1: sigmoid
__device__ __forceinline__ void gemm_hid_to_z(const float* src, const unsigned int* wf,
                                              const float* bias, float* dst,
                                              int rbase, int lane, int mode) {
  v16h ah[4];
#pragma unroll
  for (int kt = 0; kt < 4; ++kt)
    ah[kt] = packA_lds(src, HIDN, rbase, kt * 32, lane, false);
#pragma unroll
  for (int nt = 0; nt < 2; ++nt) {
    v8f c = {};
#pragma unroll
    for (int kt = 0; kt < 4; ++kt)
      c = wmma_f16(ah[kt], loadB(wf, kt * 2 + nt, lane), c);
    int nl = lane & 15, mh = (lane >> 4) << 3;
    float bv = bias[nt * 16 + nl];
#pragma unroll
    for (int v = 0; v < 8; ++v) {
      float x = c[v] + bv;
      dst[(rbase + mh + v) * ZDIM + nt * 16 + nl] =
          (mode == 0) ? x : 1.f / (1.f + expf(-x));
    }
  }
}

// ---------------- persistent kernel: one block = 1 batch column x 64 samples ----------------
__global__ void __launch_bounds__(NTHREADS) pf_scan_kernel(
    const float* __restrict__ input, const float* __restrict__ prices,
    const float* __restrict__ Wg1, const float* __restrict__ bg1,
    const float* __restrict__ Wg2, const float* __restrict__ bg2,
    const float* __restrict__ Wm1, const float* __restrict__ bm1,
    const float* __restrict__ Wm2, const float* __restrict__ bm2,
    const float* __restrict__ Wlin, const float* __restrict__ Wscale,
    const float* __restrict__ bscale, const float* __restrict__ We,
    const float* __restrict__ be, float* __restrict__ out_loss) {
  extern __shared__ char smem[];
  unsigned int* wg1f  = (unsigned int*)(smem + WG1_OFF);
  unsigned int* wm1f  = (unsigned int*)(smem + WM1_OFF);
  unsigned int* wg2f  = (unsigned int*)(smem + WG2_OFF);
  unsigned int* wm2f  = (unsigned int*)(smem + WM2_OFF);
  unsigned int* wlinf = (unsigned int*)(smem + WLIN_OFF);
  unsigned int* wscf  = (unsigned int*)(smem + WSC_OFF);
  unsigned int* wef   = (unsigned int*)(smem + WE_OFF);
  float* zprev = (float*)(smem + ZPREV_OFF);
  float* h1    = (float*)(smem + H1_OFF);
  float* gate  = (float*)(smem + GATE_OFF);
  float* prop  = (float*)(smem + PROP_OFF);
  float* locb  = (float*)(smem + LOC_OFF);
  float* sclb  = (float*)(smem + SCL_OFF);
  float* outb  = (float*)(smem + OUTB_OFF);
  float* red   = (float*)(smem + RED_OFF);
  float* baseb = (float*)(smem + BASE_OFF);

  const int tid = threadIdx.x;
  const int wg = blockIdx.x;        // batch column b
  const int lane = tid & 31;
  const int rbase = (tid >> 5) << 4; // row tile base per wave: 0,16,32,48

  // one-time: pack weights as f16 B-fragments into LDS; zero z carry
  pack_weight(Wg1, 96, HIDN, wg1f, tid, NTHREADS);
  pack_weight(Wm1, 96, HIDN, wm1f, tid, NTHREADS);
  pack_weight(Wg2, HIDN, ZDIM, wg2f, tid, NTHREADS);
  pack_weight(Wm2, HIDN, ZDIM, wm2f, tid, NTHREADS);
  pack_weight(Wlin, 96, ZDIM, wlinf, tid, NTHREADS);
  pack_weight(Wscale, ZDIM, ZDIM, wscf, tid, NTHREADS);
  pack_weight(We, ZDIM, 16, wef, tid, NTHREADS);
  for (int i = tid; i < MROWS * ZDIM; i += NTHREADS) zprev[i] = 0.f;
  __syncthreads();

  // per-row carry state (thread tid<64 owns sample s = tid of column wg)
  const int s_ = tid;               // sample index (valid when tid<64)
  const int grow = s_ * B_ + wg;    // global row id = s*B + b
  float cash = 1.f, cum_z_lp = 0.f, cash_lp = 0.f, lossv = 0.f;
  int bank = 0;
  int pstate[NC] = {0, 0, 0, 0}, ptype[NC] = {0, 0, 0, 0};
  float ipv[NC] = {0, 0, 0, 0};
  float pt00[NC] = {0, 0, 0, 0}, pt01[NC] = {0, 0, 0, 0};
  float pt10[NC] = {0, 0, 0, 0}, pt11[NC] = {0, 0, 0, 0};
  float pcel[NC] = {0, 0, 0, 0}, ipvl[NC] = {0, 0, 0, 0};

  for (int t = 0; t < SEQ; ++t) {
    // prefetch next step's inputs while this step's GEMMs run
    if (t + 1 < SEQ) {
      const float* xn = input + ((t + 1) * B_ + wg) * INDIM;
      __builtin_prefetch(xn + lane * 2, 0, 1);                       // 256B row
      __builtin_prefetch(prices + (t + 1) * 2 * NC * B_ + tid * 8, 0, 1); // 4KB slab
    }

    // ---------- GEMM phase (all 4 waves, EXEC all-ones) ----------
    const float* xv = input + (t * B_ + wg) * INDIM;
    v16h a0 = packA_bc(xv, 0, lane);
    v16h a1 = packA_bc(xv, 32, lane);
    v16h a2 = packA_lds(zprev, ZDIM, rbase, 0, lane, false);

    gemm_h_to_hid(a0, a1, a2, wg1f, bg1, h1, rbase, lane);         // H1g
    gemm_hid_to_z(h1, wg2f, bg2, gate, rbase, lane, 1);            // gate
    gemm_h_to_hid(a0, a1, a2, wm1f, bm1, h1, rbase, lane);         // H1m
    gemm_hid_to_z(h1, wm2f, bm2, prop, rbase, lane, 0);            // prop

    // lin = h @ Wlin; fuse loc = (1-gate)*lin + gate*prop
#pragma unroll
    for (int nt = 0; nt < 2; ++nt) {
      v8f c = {};
      c = wmma_f16(a0, loadB(wlinf, nt, lane), c);
      c = wmma_f16(a1, loadB(wlinf, 2 + nt, lane), c);
      c = wmma_f16(a2, loadB(wlinf, 4 + nt, lane), c);
      int nl = lane & 15, mh = (lane >> 4) << 3;
#pragma unroll
      for (int v = 0; v < 8; ++v) {
        int r = rbase + mh + v, n = nt * 16 + nl;
        float g = gate[r * ZDIM + n], pp = prop[r * ZDIM + n];
        locb[r * ZDIM + n] = (1.f - g) * c[v] + g * pp;
      }
    }
    // scale = softplus(relu(prop) @ Wscale + bscale) + 1e-4
    {
      v16h ap = packA_lds(prop, ZDIM, rbase, 0, lane, true);
#pragma unroll
      for (int nt = 0; nt < 2; ++nt) {
        v8f c = {};
        c = wmma_f16(ap, loadB(wscf, nt, lane), c);
        int nl = lane & 15, mh = (lane >> 4) << 3;
        float bv = bscale[nt * 16 + nl];
#pragma unroll
        for (int v = 0; v < 8; ++v) {
          float x = c[v] + bv;
          float sp = (x > 20.f) ? x : log1pf(expf(x));
          sclb[(rbase + mh + v) * ZDIM + nt * 16 + nl] = sp + 1e-4f;
        }
      }
    }
    __syncthreads();

    // ---------- sampling phase: z = loc + scale*eps ----------
    unsigned int ksa, ksb, k1a, k1b, k2a, k2b, k3a, k3b;
    threefry2x32(0u, 42u, 0u, (unsigned int)t, ksa, ksb);  // fold_in(base, t)
    threefry2x32(ksa, ksb, 0u, 0u, k1a, k1b);              // split -> k1,k2,k3
    threefry2x32(ksa, ksb, 0u, 1u, k2a, k2b);
    threefry2x32(ksa, ksb, 0u, 2u, k3a, k3b);
    if (tid < MROWS) {
      float zlp = 0.f;
#pragma unroll
      for (int i = 0; i < 16; ++i) {
        unsigned int r0, r1;
        threefry2x32(k1a, k1b, (unsigned int)grow, (unsigned int)i, r0, r1);
        float uA = fmaxf(u01(r0), 1e-7f);
        float uB = u01(r1);
        float rad = sqrtf(-2.f * logf(uA));
        float ang = 6.28318530718f * uB;
        float e0 = rad * cosf(ang), e1 = rad * sinf(ang);
        int d0 = 2 * i, d1 = 2 * i + 1;
        float l0 = locb[tid * ZDIM + d0], sc0 = sclb[tid * ZDIM + d0];
        float l1 = locb[tid * ZDIM + d1], sc1 = sclb[tid * ZDIM + d1];
        zprev[tid * ZDIM + d0] = l0 + sc0 * e0;
        zprev[tid * ZDIM + d1] = l1 + sc1 * e1;
        zlp += -0.5f * (e0 * e0 + e1 * e1) - logf(sc0) - logf(sc1) - LOG2PI_;
      }
      cum_z_lp += zlp;
    }
    __syncthreads();

    // ---------- out = sigmoid(z @ We + be) ----------
    {
      v16h az = packA_lds(zprev, ZDIM, rbase, 0, lane, false);
      v8f c = {};
      c = wmma_f16(az, loadB(wef, 0, lane), c);
      int nl = lane & 15, mh = (lane >> 4) << 3;
      float bv = be[nl];
#pragma unroll
      for (int v = 0; v < 8; ++v) {
        float x = c[v] + bv;
        outb[(rbase + mh + v) * 16 + nl] = 1.f / (1.f + expf(-x));
      }
    }
    __syncthreads();

    // ---------- trading bookkeeping ----------
    float costs[NC];
    int open_mask[NC], close_mask[NC];
    if (tid < MROWS) {
      float pos_pl[NC];
      int open_pos[NC], closed_pos[NC];
      float total_pos = 0.f;
      int any_open = 0;
#pragma unroll
      for (int j = 0; j < NC; ++j) {
        float p0 = prices[((t * 2 + 0) * NC + j) * B_ + wg];
        float p1 = prices[((t * 2 + 1) * NC + j) * B_ + wg];
        int op = (pstate[j] == 1), cl = (pstate[j] == 0);
        open_pos[j] = op; closed_pos[j] = cl;
        int lm = (ptype[j] == 0);
        if (op) pt10[j] = 0.f;
        float ntv = 1.f / (lm ? p1 : p0);
        if (op) pt11[j] = ntv;
        float pl = op ? ipv[j] * (pt00[j] + pt10[j]) * (pt01[j] + pt11[j]) : 0.f;
        pos_pl[j] = pl;
        total_pos += op ? (ipv[j] + pl) : 0.f;
        any_open |= op;
      }
      float portfolio = cash + total_pos;
      if (any_open) bank = (portfolio <= 0.f) ? 1 : 0;

      // two currencies per Threefry call (use both 32-bit halves)
      unsigned int ue0, ue1, ue2, ue3, ut0, ut1, ut2, ut3;
      threefry2x32(k2a, k2b, (unsigned int)grow, 0u, ue0, ue1);
      threefry2x32(k2a, k2b, (unsigned int)grow, 1u, ue2, ue3);
      threefry2x32(k3a, k3b, (unsigned int)grow, 0u, ut0, ut1);
      threefry2x32(k3a, k3b, (unsigned int)grow, 1u, ut2, ut3);
      unsigned int uex[NC] = {ue0, ue1, ue2, ue3};
      unsigned int uty[NC] = {ut0, ut1, ut2, ut3};

#pragma unroll
      for (int j = 0; j < NC; ++j) {
        float p0 = prices[((t * 2 + 0) * NC + j) * B_ + wg];
        float p1 = prices[((t * 2 + 1) * NC + j) * B_ + wg];
        float op_p = bank ? 0.f : outb[tid * 16 + j];
        float cl_p = bank ? 1.f : outb[tid * 16 + 4 + j];
        float sh_p = outb[tid * 16 + 8 + j];
        float ex_p = closed_pos[j] ? op_p : cl_p;
        float u = u01(uex[j]);
        float ex_s = (u < ex_p) ? 1.f : 0.f;
        pcel[j] += bern_lp(ex_s, ex_p);
        int event = ex_s > 0.5f;
        int om = closed_pos[j] && event;
        int cm = open_pos[j] && event;
        open_mask[j] = om; close_mask[j] = cm;
        if (event) pstate[j] = (pstate[j] + 1) & 1;
        float u2v = u01(uty[j]);
        float otype = (u2v < sh_p) ? 1.f : 0.f;
        if (om) {
          pcel[j] += bern_lp(otype, sh_p);
          ptype[j] = (int)otype;
        }
        int lm = (ptype[j] == 0);
        if (om) pt00[j] = 1.0f * (lm ? p0 : -p1);     // LEVERAGE = 1
        if (open_pos[j]) pt01[j] = 1.f / (lm ? p0 : p1);
        costs[j] = cm ? pos_pl[j] : 0.f;
        red[j * MROWS + tid] = costs[j];
      }
    }
    __syncthreads();
    if (tid < NC) {  // per-currency mean over the 64 samples of this column
      float ssum = 0.f;
      for (int q = 0; q < MROWS; ++q) ssum += red[tid * MROWS + q];
      baseb[tid] = ssum * (1.f / (float)MROWS);
    }
    __syncthreads();
    if (tid < MROWS) {
#pragma unroll
      for (int j = 0; j < NC; ++j) {  // sequential over currencies (cash coupling)
        int om = open_mask[j], cm = close_mask[j];
        float frac = outb[tid * 16 + 12 + j];
        float new_ipv = frac * cash;
        if (om) {
          ipv[j] = new_ipv;
          cash -= new_ipv;
          cash_lp += pcel[j];
          ipvl[j] = cash_lp;
          pcel[j] = 0.f;
        }
        float cost = costs[j];
        float baseline = baseb[j];
        float cost_lp = cum_z_lp + ipvl[j] + pcel[j];
        if (cm) {
          lossv += cost_lp * (cost - baseline) + cost;
          cash += ipv[j] + cost;
          cash_lp += pcel[j];
          pcel[j] = 0.f;
        }
      }
    }
    __syncthreads();
  }
  if (tid < MROWS) out_loss[s_ * B_ + wg] = lossv;
}

extern "C" void kernel_launch(void* const* d_in, const int* in_sizes, int n_in,
                              void* d_out, int out_size, void* d_ws, size_t ws_size,
                              hipStream_t stream) {
  (void)in_sizes; (void)n_in; (void)out_size; (void)d_ws; (void)ws_size;
  const float* input  = (const float*)d_in[0];
  const float* prices = (const float*)d_in[1];
  const float* Wg1 = (const float*)d_in[2];
  const float* bg1 = (const float*)d_in[3];
  const float* Wg2 = (const float*)d_in[4];
  const float* bg2 = (const float*)d_in[5];
  const float* Wm1 = (const float*)d_in[6];
  const float* bm1 = (const float*)d_in[7];
  const float* Wm2 = (const float*)d_in[8];
  const float* bm2 = (const float*)d_in[9];
  const float* Wlin = (const float*)d_in[10];
  const float* Wscale = (const float*)d_in[11];
  const float* bscale = (const float*)d_in[12];
  const float* We = (const float*)d_in[13];
  const float* be = (const float*)d_in[14];
  float* out = (float*)d_out;

  static_assert(SMEM_BYTES <= 160 * 1024, "want 2 blocks per WGP (320KB LDS)");
  hipFuncSetAttribute((const void*)pf_scan_kernel,
                      hipFuncAttributeMaxDynamicSharedMemorySize, SMEM_BYTES);
  pf_scan_kernel<<<dim3(B_), dim3(NTHREADS), SMEM_BYTES, stream>>>(
      input, prices, Wg1, bg1, Wg2, bg2, Wm1, bm1, Wm2, bm2, Wlin, Wscale,
      bscale, We, be, out);
}
